// AttnBlock_70368744177938
// MI455X (gfx1250) — matrix operand order repaired and checked by reading, not compiled, not run on hardware
//
#include <hip/hip_runtime.h>

typedef __attribute__((ext_vector_type(16))) __bf16 v16bf;
typedef __attribute__((ext_vector_type(8)))  __bf16 v8bf;
typedef __attribute__((ext_vector_type(8)))  float  v8f;

#define CH     512
#define NPIX   4096
#define BATCH  4
#define GROUPS 32
#define CPG    16                      // CH / GROUPS
#define SCALE  0.044194173824159216f   // 512^-0.5
#define EPS    1e-6f

// ---- WMMA helpers -----------------------------------------------------------
// 16-bit A-matrix 16x32 fragment in this code's contiguous k order (lane l<16: row l, K 0..15;
// lane>=16: row l-16, K 16..31) loaded from row-major [M][K] memory.
static __device__ __forceinline__ v16bf load_a_frag(const __bf16* row, int c0, int lhi) {
  return *(const v16bf*)(row + c0 + (lhi ? 16 : 0));
}
// B-matrix 32x16 fragment (lane l<16: col l, K 0..15; lane>=16: col l-16,
// K 16..31) loaded from column-major storage (K contiguous per column).
static __device__ __forceinline__ v16bf load_b_frag(const __bf16* col, int c0, int lhi) {
  return *(const v16bf*)(col + c0 + (lhi ? 16 : 0));
}
static __device__ __forceinline__ v8f wmma_bf16(v16bf a, v16bf b, v8f c) {
  return __builtin_amdgcn_wmma_f32_16x16x32_bf16(false, a, false, b, (short)0, c, false, false);
}

// ---- fp32 -> bf16 weight convert -------------------------------------------
__global__ __launch_bounds__(256) void cvt_kernel(const float* __restrict__ w,
                                                  __bf16* __restrict__ o, int n) {
  int i = blockIdx.x * 256 + threadIdx.x;
  if (i < n) o[i] = (__bf16)w[i];
}

// ---- GroupNorm stats: one block per (b, group) ------------------------------
__global__ __launch_bounds__(256) void gn_stats_kernel(const float* __restrict__ x,
                                                       float* __restrict__ stats) {
  __shared__ float ss[256], sq[256];
  const int b = blockIdx.x >> 5;
  const int g = blockIdx.x & 31;
  const float* p = x + ((size_t)b * CH + (size_t)g * CPG) * NPIX;
  const int n = CPG * NPIX; // 65536
  float s = 0.f, q = 0.f;
  for (int i = threadIdx.x; i < n; i += 256) { float v = p[i]; s += v; q += v * v; }
  ss[threadIdx.x] = s; sq[threadIdx.x] = q;
  __syncthreads();
  for (int off = 128; off > 0; off >>= 1) {
    if (threadIdx.x < off) {
      ss[threadIdx.x] += ss[threadIdx.x + off];
      sq[threadIdx.x] += sq[threadIdx.x + off];
    }
    __syncthreads();
  }
  if (threadIdx.x == 0) {
    float mean = ss[0] / n;
    float var  = sq[0] / n - mean * mean;
    stats[2 * blockIdx.x]     = mean;
    stats[2 * blockIdx.x + 1] = rsqrtf(var + EPS);
  }
}

// ---- normalize + transpose to [b][n][c] bf16 --------------------------------
__global__ __launch_bounds__(256) void gn_apply_kernel(const float* __restrict__ x,
    const float* __restrict__ gamma, const float* __restrict__ beta,
    const float* __restrict__ stats, __bf16* __restrict__ hnT) {
  const int idx = blockIdx.x * 256 + threadIdx.x;   // b*N*C + n*C + c
  const int c = idx & (CH - 1);
  const int n = (idx >> 9) & (NPIX - 1);
  const int b = idx >> 21;
  const int sg = 2 * (b * GROUPS + (c >> 4));
  const float mean = stats[sg], rstd = stats[sg + 1];
  const float xv = x[((size_t)b * CH + c) * NPIX + n];
  hnT[idx] = (__bf16)((xv - mean) * rstd * gamma[c] + beta[c]);
}

// ---- fused Q/K/V 1x1-conv GEMM: shares B fragment across 3 WMMAs ------------
// q,k stored [b][n][c]; v stored [b][c][n] (so attention B-frags are contiguous)
__global__ __launch_bounds__(256) void qkv_kernel(
    const __bf16* __restrict__ hnT,
    const __bf16* __restrict__ wqb, const __bf16* __restrict__ wkb,
    const __bf16* __restrict__ wvb,
    const float* __restrict__ bq, const float* __restrict__ bk,
    const float* __restrict__ bv,
    __bf16* __restrict__ qT, __bf16* __restrict__ kT, __bf16* __restrict__ vC) {
  const int tid  = threadIdx.x;
  const int lane = tid & 31;
  const int lmod = lane & 15, lhi = lane >> 4;
  const int wg  = blockIdx.x * 8 + (tid >> 5);
  const int b   = wg >> 13;          // / (32*256)
  const int rem = wg & 8191;
  const int o0  = (rem >> 8) * 16;
  const int n0  = (rem & 255) * 16;
  const size_t base = (size_t)b * NPIX * CH;

  const __bf16* brow = hnT + base + (size_t)(n0 + lmod) * CH;
  const __bf16* aq = wqb + (size_t)(o0 + lmod) * CH;
  const __bf16* ak = wkb + (size_t)(o0 + lmod) * CH;
  const __bf16* av = wvb + (size_t)(o0 + lmod) * CH;

  v8f accq = {}, acck = {}, accv = {};
  for (int c0 = 0; c0 < CH; c0 += 32) {
    v16bf bf = load_b_frag(brow, c0, lhi);
    accq = wmma_bf16(load_a_frag(aq, c0, lhi), bf, accq);
    acck = wmma_bf16(load_a_frag(ak, c0, lhi), bf, acck);
    accv = wmma_bf16(load_a_frag(av, c0, lhi), bf, accv);
  }
  // D layout: lane<16 VGPR j -> (M=j, N=lane); lane>=16 -> (M=8+j, N=lane-16)
  const int ob = o0 + 8 * lhi;
  const int n  = n0 + lmod;
  v8bf qv, kv;
#pragma unroll
  for (int j = 0; j < 8; j++) {
    qv[j] = (__bf16)(accq[j] + bq[ob + j]);
    kv[j] = (__bf16)(acck[j] + bk[ob + j]);
  }
  *(v8bf*)(qT + base + (size_t)n * CH + ob) = qv;
  *(v8bf*)(kT + base + (size_t)n * CH + ob) = kv;
#pragma unroll
  for (int j = 0; j < 8; j++)
    vC[base + (size_t)(ob + j) * NPIX + n] = (__bf16)(accv[j] + bv[ob + j]);
}

// ---- flash attention: 64 query rows per block, m-tiles of 64 ----------------
__global__ __launch_bounds__(512) void attn_kernel(
    const __bf16* __restrict__ qT, const __bf16* __restrict__ kT,
    const __bf16* __restrict__ vC, __bf16* __restrict__ Obuf) {
  __shared__ float sS[64][68];                     // padded vs bank conflicts
  __shared__ __align__(32) __bf16 sP[64][64];
  __shared__ float row_m[64], row_l[64], row_corr[64];

  const int tid  = threadIdx.x;
  const int lane = tid & 31, wave = tid >> 5;
  const int lmod = lane & 15, lhi = lane >> 4;
  const int r = wave >> 2;                          // query row sub-block 0..3
  const int cg = wave & 3;                          // output-channel group 0..3
  const int b  = blockIdx.x >> 6;
  const int n0 = (blockIdx.x & 63) * 64;
  const size_t base = (size_t)b * NPIX * CH;

  v8f acc[8] = {};                                  // 16 rows x 128 cols f32

  if (tid < 64) { row_m[tid] = -3.0e38f; row_l[tid] = 0.f; }
  __syncthreads();

  const __bf16* qrow = qT + base + (size_t)(n0 + 16 * r + lmod) * CH;
  const int myrow = 16 * r + 8 * lhi;               // + j

  for (int m0 = 0; m0 < NPIX; m0 += 64) {
    // S tile: each wave one 16x16 tile of the 64x64 score block
    const __bf16* krow = kT + base + (size_t)(m0 + 16 * cg + lmod) * CH;
    v8f s = {};
    for (int c0 = 0; c0 < CH; c0 += 32)
      s = wmma_bf16(load_a_frag(qrow, c0, lhi), load_b_frag(krow, c0, lhi), s);
#pragma unroll
    for (int j = 0; j < 8; j++)
      sS[myrow + j][16 * cg + lmod] = s[j] * SCALE;
    __syncthreads();

    // online softmax: one thread per query row
    if (tid < 64) {
      float mold = row_m[tid], mnew = mold;
      for (int j = 0; j < 64; j++) mnew = fmaxf(mnew, sS[tid][j]);
      float corr = __expf(mold - mnew);
      float lsum = 0.f;
      for (int j = 0; j < 64; j++) {
        float p = __expf(sS[tid][j] - mnew);
        lsum += p;
        sP[tid][j] = (__bf16)p;
      }
      row_corr[tid] = corr;
      row_m[tid] = mnew;
      row_l[tid] = row_l[tid] * corr + lsum;
    }
    __syncthreads();

    // rescale accumulators, then O += P @ V^T
    float cj[8];
#pragma unroll
    for (int j = 0; j < 8; j++) cj[j] = row_corr[myrow + j];
#pragma unroll
    for (int t = 0; t < 8; t++)
#pragma unroll
      for (int j = 0; j < 8; j++) acc[t][j] *= cj[j];

#pragma unroll
    for (int k0 = 0; k0 < 64; k0 += 32) {
      v16bf a = load_a_frag(&sP[16 * r + lmod][0], k0, lhi);
#pragma unroll
      for (int t = 0; t < 8; t++) {
        const __bf16* vp = vC + base
            + (size_t)(128 * cg + 16 * t + lmod) * NPIX + m0 + k0 + 16 * lhi;
        acc[t] = wmma_bf16(a, *(const v16bf*)vp, acc[t]);
      }
    }
    __syncthreads();
  }

  float invl[8];
#pragma unroll
  for (int j = 0; j < 8; j++) invl[j] = 1.f / row_l[myrow + j];
#pragma unroll
  for (int t = 0; t < 8; t++)
#pragma unroll
    for (int j = 0; j < 8; j++)
      Obuf[base + (size_t)(n0 + myrow + j) * CH + 128 * cg + 16 * t + lmod] =
          (__bf16)(acc[t][j] * invl[j]);
}

// ---- output projection + bias + residual (fp32 out) -------------------------
__global__ __launch_bounds__(256) void outproj_kernel(
    const __bf16* __restrict__ Obuf, const __bf16* __restrict__ wob,
    const float* __restrict__ bo, const float* __restrict__ x,
    float* __restrict__ out) {
  const int tid  = threadIdx.x;
  const int lane = tid & 31;
  const int lmod = lane & 15, lhi = lane >> 4;
  const int wg  = blockIdx.x * 8 + (tid >> 5);
  const int b   = wg >> 13;
  const int rem = wg & 8191;
  const int o0  = (rem >> 8) * 16;
  const int n0  = (rem & 255) * 16;
  const size_t base = (size_t)b * NPIX * CH;

  const __bf16* brow = Obuf + base + (size_t)(n0 + lmod) * CH;
  const __bf16* arow = wob + (size_t)(o0 + lmod) * CH;
  v8f acc = {};
  for (int c0 = 0; c0 < CH; c0 += 32)
    acc = wmma_bf16(load_a_frag(arow, c0, lhi), load_b_frag(brow, c0, lhi), acc);

#pragma unroll
  for (int j = 0; j < 8; j++) {
    const int o = o0 + 8 * lhi + j;
    const size_t oi = ((size_t)b * CH + o) * NPIX + n0 + lmod;
    out[oi] = x[oi] + acc[j] + bo[o];
  }
}

// ---- host launch ------------------------------------------------------------
extern "C" void kernel_launch(void* const* d_in, const int* in_sizes, int n_in,
                              void* d_out, int out_size, void* d_ws, size_t ws_size,
                              hipStream_t stream) {
  const float* x     = (const float*)d_in[0];
  const float* gamma = (const float*)d_in[1];
  const float* beta  = (const float*)d_in[2];
  const float* wq = (const float*)d_in[3]; const float* bq = (const float*)d_in[4];
  const float* wk = (const float*)d_in[5]; const float* bk = (const float*)d_in[6];
  const float* wv = (const float*)d_in[7]; const float* bv = (const float*)d_in[8];
  const float* wo = (const float*)d_in[9]; const float* bo = (const float*)d_in[10];
  float* out = (float*)d_out;

  // workspace: stats(1KB) | 4 bf16 weights (2MB) | hnT/O | qT | kT | v (~67MB)
  char* ws = (char*)d_ws;
  float* stats = (float*)ws;
  size_t off = 1024;
  const size_t wsz = (size_t)CH * CH * sizeof(__bf16);
  __bf16* wqb = (__bf16*)(ws + off); off += wsz;
  __bf16* wkb = (__bf16*)(ws + off); off += wsz;
  __bf16* wvb = (__bf16*)(ws + off); off += wsz;
  __bf16* wob = (__bf16*)(ws + off); off += wsz;
  const size_t big = (size_t)BATCH * NPIX * CH * sizeof(__bf16);
  __bf16* hnT = (__bf16*)(ws + off); off += big;   // reused as attention O
  __bf16* qT  = (__bf16*)(ws + off); off += big;
  __bf16* kT  = (__bf16*)(ws + off); off += big;
  __bf16* vC  = (__bf16*)(ws + off); off += big;
  __bf16* Obuf = hnT;

  const int wElems = CH * CH;
  cvt_kernel<<<(wElems + 255) / 256, 256, 0, stream>>>(wq, wqb, wElems);
  cvt_kernel<<<(wElems + 255) / 256, 256, 0, stream>>>(wk, wkb, wElems);
  cvt_kernel<<<(wElems + 255) / 256, 256, 0, stream>>>(wv, wvb, wElems);
  cvt_kernel<<<(wElems + 255) / 256, 256, 0, stream>>>(wo, wob, wElems);

  gn_stats_kernel<<<BATCH * GROUPS, 256, 0, stream>>>(x, stats);
  gn_apply_kernel<<<(BATCH * NPIX * CH) / 256, 256, 0, stream>>>(x, gamma, beta, stats, hnT);
  qkv_kernel<<<4096, 256, 0, stream>>>(hnT, wqb, wkb, wvb, bq, bk, bv, qT, kT, vC);
  attn_kernel<<<BATCH * 64, 512, 0, stream>>>(qT, kT, vC, Obuf);
  outproj_kernel<<<4096, 256, 0, stream>>>(Obuf, wob, bo, x, out);
}
